// MOEgate_88338887344193
// MI455X (gfx1250) — matrix operand order repaired and checked by reading, not compiled, not run on hardware
//
#include <hip/hip_runtime.h>

// ---------------- problem constants (from reference setup) ----------------
#define DIM    1024   // D
#define EXP    64     // E experts
#define BATCH  4      // B
#define TOPK   8
#define KSTEP  32     // WMMA K for f16
#define WG_TOK 256    // tokens per workgroup (8 waves x 2 M-tiles x 16)
#define SWH    (DIM + 8)   // LDS W row stride in halves (2064B: 16B pad kills bank conflicts)
#define LROW   (EXP + 1)   // logits/scores LDS row stride in floats (65 -> conflict-free)

typedef __attribute__((ext_vector_type(16))) _Float16 v16h;
typedef __attribute__((ext_vector_type(8)))  _Float16 v8h;
typedef __attribute__((ext_vector_type(8)))  float    v8f;

// --------------------------------------------------------------------------
// Zero the [B,E] score-sum and [B,E] ce accumulators in workspace (512 f32).
__global__ void moe_init_ws_kernel(float* __restrict__ ws) {
    ws[threadIdx.x] = 0.0f;   // launched with 512 threads
}

// --------------------------------------------------------------------------
// Main fused kernel: GEMM (f16 WMMA, fp32 accum) + softmax + top-8 + partial aux.
__global__ void moe_gate_kernel(const float* __restrict__ x,
                                const float* __restrict__ Wm,
                                float* __restrict__ idx_out,
                                float* __restrict__ w_out,
                                float* __restrict__ ws_score,
                                float* __restrict__ ws_ce,
                                int S) {
    extern __shared__ char smem[];
    _Float16* w_lds  = (_Float16*)smem;   // phase 0/1: W as f16, [EXP][SWH]
    float*    sc_lds = (float*)smem;      // phase 2/3: logits->scores, [256][LROW]
    __shared__ float ce_lds[EXP];

    const int tid  = threadIdx.x;
    const int wid  = tid >> 5;
    const int lane = tid & 31;
    const int m    = lane & 15;   // A row / B col within 16x16 tile
    const int h    = lane >> 4;   // K-half selector

    if (tid < EXP) ce_lds[tid] = 0.0f;

    // ---- Phase 0: stage W (fp32 global) -> fp16 LDS with padded rows ----
    for (int i = tid * 8; i < EXP * DIM; i += 256 * 8) {
        const int e = i >> 10;           // /DIM
        const int k = i & (DIM - 1);
        const float4 f0 = *(const float4*)(Wm + e * DIM + k);
        const float4 f1 = *(const float4*)(Wm + e * DIM + k + 4);
        v8h hv;
        hv[0] = (_Float16)f0.x; hv[1] = (_Float16)f0.y;
        hv[2] = (_Float16)f0.z; hv[3] = (_Float16)f0.w;
        hv[4] = (_Float16)f1.x; hv[5] = (_Float16)f1.y;
        hv[6] = (_Float16)f1.z; hv[7] = (_Float16)f1.w;
        *(v8h*)(w_lds + e * SWH + k) = hv;   // ds_store_b128, 16B aligned
    }
    __syncthreads();

    // ---- Phase 1: GEMM. Wave owns 32 tokens (2 M-tiles) x 64 experts (4 N-tiles)
    v8f acc[2][4];
    const v8f vzero = {0.f, 0.f, 0.f, 0.f, 0.f, 0.f, 0.f, 0.f};
#pragma unroll
    for (int mt = 0; mt < 2; ++mt)
#pragma unroll
        for (int nt = 0; nt < 4; ++nt) acc[mt][nt] = vzero;

    const long t0  = (long)blockIdx.x * WG_TOK + wid * 32;
    const float* xr0 = x + (t0 + m) * (long)DIM;
    const float* xr1 = x + (t0 + 16 + m) * (long)DIM;

    for (int kb = 0; kb < DIM; kb += KSTEP) {
        // A fragment (ISA 16-bit A 16x32 layout): lane half h holds K = kb + {8h..8h+7, 16+8h..23+8h}
        v16h afr[2];
#pragma unroll
        for (int mt = 0; mt < 2; ++mt) {
            const float* p = (mt ? xr1 : xr0) + kb + 8 * h;
            const float4 f0 = *(const float4*)(p);
            const float4 f1 = *(const float4*)(p + 4);
            const float4 f2 = *(const float4*)(p + 16);
            const float4 f3 = *(const float4*)(p + 20);
            v16h a;
            a[0]  = (_Float16)f0.x; a[1]  = (_Float16)f0.y;
            a[2]  = (_Float16)f0.z; a[3]  = (_Float16)f0.w;
            a[4]  = (_Float16)f1.x; a[5]  = (_Float16)f1.y;
            a[6]  = (_Float16)f1.z; a[7]  = (_Float16)f1.w;
            a[8]  = (_Float16)f2.x; a[9]  = (_Float16)f2.y;
            a[10] = (_Float16)f2.z; a[11] = (_Float16)f2.w;
            a[12] = (_Float16)f3.x; a[13] = (_Float16)f3.y;
            a[14] = (_Float16)f3.z; a[15] = (_Float16)f3.w;
            afr[mt] = a;
        }
        // B fragment (16-bit B KxN layout): lane half h holds K = kb + {8h..8h+7, 16+8h..23+8h}
        v16h bfr[4];
#pragma unroll
        for (int nt = 0; nt < 4; ++nt) {
            const _Float16* q = w_lds + (nt * 16 + m) * SWH + kb + 8 * h;
            const v8h lo = *(const v8h*)(q);       // ds_load_b128
            const v8h hi = *(const v8h*)(q + 16);  // ds_load_b128
            bfr[nt] = __builtin_shufflevector(lo, hi, 0,1,2,3,4,5,6,7,8,9,10,11,12,13,14,15);
        }
#pragma unroll
        for (int mt = 0; mt < 2; ++mt)
#pragma unroll
            for (int nt = 0; nt < 4; ++nt)
                acc[mt][nt] = __builtin_amdgcn_wmma_f32_16x16x32_f16(
                    false, afr[mt], false, bfr[nt], (short)0, acc[mt][nt], false, false);
    }

    // ---- Phase 1.5: dump logits to LDS (reuse W region; barrier first) ----
    __syncthreads();
#pragma unroll
    for (int mt = 0; mt < 2; ++mt)
#pragma unroll
        for (int nt = 0; nt < 4; ++nt)
#pragma unroll
            for (int g = 0; g < 8; ++g) {
                const int r = wid * 32 + mt * 16 + h * 8 + g;   // C layout: M = g + 8h
                sc_lds[r * LROW + nt * 16 + m] = acc[mt][nt][g];
            }
    __syncthreads();

    // ---- Phase 2: one token per thread: softmax, top-8, renorm, outputs ----
    {
        const int  r = tid;
        const long t = (long)blockIdx.x * WG_TOK + r;
        float s[EXP];
#pragma unroll
        for (int e = 0; e < EXP; ++e) s[e] = sc_lds[r * LROW + e];
        float mx = s[0];
#pragma unroll
        for (int e = 1; e < EXP; ++e) mx = fmaxf(mx, s[e]);
        float sum = 0.f;
#pragma unroll
        for (int e = 0; e < EXP; ++e) { s[e] = __expf(s[e] - mx); sum += s[e]; }
        const float inv = 1.0f / sum;
#pragma unroll
        for (int e = 0; e < EXP; ++e) { s[e] *= inv; sc_lds[r * LROW + e] = s[e]; }

        unsigned long long chosen = 0ull;
        float wsum = 0.f;
        float tw[TOPK]; int ti[TOPK];
#pragma unroll
        for (int kk = 0; kk < TOPK; ++kk) {
            float best = -1.0f; int bi = 0;
#pragma unroll
            for (int e = 0; e < EXP; ++e) {
                const bool take = (((chosen >> e) & 1ull) == 0ull) && (s[e] > best);
                best = take ? s[e] : best;
                bi   = take ? e    : bi;
            }
            chosen |= (1ull << bi);
            wsum  += best;
            tw[kk] = best; ti[kk] = bi;
            atomicAdd(&ce_lds[bi], 1.0f);
        }
        const float rn = 1.0f / (wsum + 1e-20f);
#pragma unroll
        for (int kk = 0; kk < TOPK; ++kk) {
            idx_out[t * TOPK + kk] = (float)ti[kk];
            w_out[t * TOPK + kk]   = tw[kk] * rn;
        }
    }
    __syncthreads();

    // ---- Phase 3: per-block expert score sums + ce -> global accumulators ----
    if (tid < EXP) {
        float a = 0.f;
        for (int r2 = 0; r2 < WG_TOK; ++r2) a += sc_lds[r2 * LROW + tid];
        const int b = (int)(((long)blockIdx.x * WG_TOK) / S);   // block lies in one batch
        atomicAdd(&ws_score[b * EXP + tid], a);
        atomicAdd(&ws_ce[b * EXP + tid], ce_lds[tid]);
    }
}

// --------------------------------------------------------------------------
// aux_loss = ALPHA * (1/B) * sum_{b,e} ce[b,e] * score_sum[b,e] / S
__global__ void moe_finalize_kernel(const float* __restrict__ ws_score,
                                    const float* __restrict__ ws_ce,
                                    float* __restrict__ aux_out, int S) {
    __shared__ float red[EXP];
    const int e = threadIdx.x;
    float p = 0.f;
#pragma unroll
    for (int b = 0; b < BATCH; ++b) p += ws_ce[b * EXP + e] * ws_score[b * EXP + e];
    red[e] = p;
    __syncthreads();
    if (e == 0) {
        float tot = 0.f;
#pragma unroll
        for (int i = 0; i < EXP; ++i) tot += red[i];
        aux_out[0] = 0.1f * tot / ((float)S * (float)BATCH);
    }
}

// --------------------------------------------------------------------------
extern "C" void kernel_launch(void* const* d_in, const int* in_sizes, int n_in,
                              void* d_out, int out_size, void* d_ws, size_t ws_size,
                              hipStream_t stream) {
    const float* x  = (const float*)d_in[0];   // [B*S, D] fp32
    const float* Wm = (const float*)d_in[1];   // [E, D]   fp32
    const int T = in_sizes[0] / DIM;           // 32768 tokens
    const int S = T / BATCH;                   // 8192

    float* out     = (float*)d_out;
    float* idx_out = out;                        // [T,8] indices (as float)
    float* w_out   = out + (size_t)T * TOPK;     // [T,8] weights
    float* aux_out = out + (size_t)2 * T * TOPK; // scalar

    float* ws       = (float*)d_ws;
    float* ws_score = ws;                // [B,E]
    float* ws_ce    = ws + BATCH * EXP;  // [B,E]

    hipLaunchKernelGGL(moe_init_ws_kernel, dim3(1), dim3(2 * BATCH * EXP), 0, stream, ws);

    const int    blocks = T / WG_TOK;                            // 128
    const size_t lds    = (size_t)EXP * SWH * sizeof(_Float16);  // 132096 B (> 256*LROW*4 scores)
    hipLaunchKernelGGL(moe_gate_kernel, dim3(blocks), dim3(256), lds, stream,
                       x, Wm, idx_out, w_out, ws_score, ws_ce, S);

    hipLaunchKernelGGL(moe_finalize_kernel, dim3(1), dim3(EXP), 0, stream,
                       ws_score, ws_ce, aux_out, S);
}
